// HadamardBlock_13142599926314
// MI455X (gfx1250) — matrix operand order repaired and checked by reading, not compiled, not run on hardware
//
#include <hip/hip_runtime.h>

#define EMB 128
#define NATOMS 50000
#define NEDGES 800000
#define ASTRIPS (NATOMS / 16)  // 3125 (exact)
#define ESTRIPS (NEDGES / 16)  // 50000 (exact)
#define TSTRIDE 136            // padded f32 stride for transpose strip (16B-aligned, bank-spread)
#define WPB 4                  // waves per block (wave32)
#define BLOCK (WPB * 32)
#define AGRID 512              // atom-side blocks (grid-stride)
#define EGRID 2048             // edge-side blocks (grid-stride)

#define SILU_SCALE 1.6666666666666667f  // 1/0.6
#define INV_SQRT2 0.7071067811865476f

typedef __attribute__((ext_vector_type(16))) __bf16 v16bf;
typedef __attribute__((ext_vector_type(8))) float v8f;
typedef __attribute__((ext_vector_type(4))) float v4f;

__device__ __forceinline__ v8f wmma_bf16(v16bf a, v16bf b, v8f c) {
  // D = A(16x32 bf16) x B(32x16 bf16) + C(16x16 f32)
  return __builtin_amdgcn_wmma_f32_16x16x32_bf16(false, a, false, b, (short)0, c,
                                                 false, false);
}

__device__ __forceinline__ v8f zero8() {
  v8f z = {0.f, 0.f, 0.f, 0.f, 0.f, 0.f, 0.f, 0.f};
  return z;
}

__device__ __forceinline__ v8f scaled_silu8(v8f x) {
  v8f r;
#pragma unroll
  for (int i = 0; i < 8; ++i) {
    float v = x[i];
    r[i] = (v / (1.0f + __expf(-v))) * SILU_SCALE;  // silu(x)*1/0.6
  }
  return r;
}

// Wave-local LDS write->read ordering (cross-lane within one wave).
__device__ __forceinline__ void wave_lds_fence() {
  __builtin_amdgcn_wave_barrier();
  asm volatile("s_wait_dscnt 0" ::: "memory");
  __builtin_amdgcn_wave_barrier();
}

// Stage a 128x128 row-major f32 weight into LDS as bf16 in B-fragment order.
// Fragment(kk, n): lane L, element j  <->  B[K][N],
//   K = kk*32 + ((L>=16)?8:0) + (j&7) + ((j>=8)?16:0) ,  N = n*16 + (L&15)
// Global reads coalesced; LDS scatter is cheap.
__device__ __forceinline__ void stage_weight(const float* __restrict__ w,
                                             __bf16* lds_w) {
  for (int t = threadIdx.x; t < EMB * EMB; t += blockDim.x) {
    int K = t >> 7;
    int c = t & 127;
    int kk = K >> 5;
    int lane = (((K >> 3) & 1) << 4) | (c & 15);
    int j = (K & 7) | (((K >> 4) & 1) << 3);
    int n = c >> 4;
    int i = (((((kk << 3) | n) << 5) | lane) << 4) | j;
    lds_w[i] = (__bf16)w[t];
  }
}

__device__ __forceinline__ v16bf ldb(const __bf16* lds_w, int kk, int n, int lane) {
  // one contiguous 32B chunk per lane -> ds_load_b128 x2
  return *(const v16bf*)(lds_w + ((((((kk << 3) | n) << 5) | lane) << 4)));
}

// A-fragment (16x32 bf16). rowp = &src[(row0 + (lane&15)) * 128].
// M = lane&15, kb = (lane>=16)?8:0; elems 0..7 -> K=k0+kb+0..7,
// elems 8..15 -> K=k0+16+kb+0..7.  Two 32B runs -> 4x v4f loads, cvt to bf16.
__device__ __forceinline__ v16bf lda_rowptr(const float* __restrict__ rowp, int k0,
                                            int lane, float scale) {
  int kb = (lane >> 4) << 3;
  const float* p = rowp + k0 + kb;
  v4f x0 = *(const v4f*)(p);
  v4f x1 = *(const v4f*)(p + 4);
  v4f x2 = *(const v4f*)(p + 16);
  v4f x3 = *(const v4f*)(p + 20);
  v16bf a;
#pragma unroll
  for (int i = 0; i < 4; ++i) {
    a[i] = (__bf16)(x0[i] * scale);
    a[4 + i] = (__bf16)(x1[i] * scale);
    a[8 + i] = (__bf16)(x2[i] * scale);
    a[12 + i] = (__bf16)(x3[i] * scale);
  }
  return a;
}

// 16x128 strip GEMM vs staged 128x128 weight: 8 N-tiles x 4 K-steps = 32 WMMAs.
__device__ __forceinline__ void gemm_acc(const v16bf a[4], const __bf16* lds_w,
                                         int lane, v8f acc[8]) {
#pragma unroll
  for (int n = 0; n < 8; ++n) {
#pragma unroll
    for (int kk = 0; kk < 4; ++kk) {
      acc[n] = wmma_bf16(a[kk], ldb(lds_w, kk, n, lane), acc[n]);
    }
  }
}

// C layout: acc[n][r], lane: M = r + 8*(lane>=16), N = lane&15, col = n*16+N.
__device__ __forceinline__ void cstore_strip(const v8f acc[8], float* strip,
                                             int lane) {
  int N = lane & 15;
  int moff = (lane >> 4) << 3;
#pragma unroll
  for (int n = 0; n < 8; ++n) {
#pragma unroll
    for (int r = 0; r < 8; ++r) {
      strip[(moff + r) * TSTRIDE + n * 16 + N] = acc[n][r];
    }
  }
}

__device__ __forceinline__ void lda_strip(const float* strip, int lane, v16bf a[4]) {
  const float* rowp = strip + (lane & 15) * TSTRIDE;
#pragma unroll
  for (int kk = 0; kk < 4; ++kk) a[kk] = lda_rowptr(rowp, kk * 32, lane, 1.0f);
}

// ---------------------------------------------------------------------------
// Residual block: xout = (xin + ssilu(ssilu(xin@w0)@w1)) * inv_sqrt2.
// Both weights staged once per block; grid-stride over 16-row strips.
__global__ void __launch_bounds__(BLOCK) k_resblock(
    const float* __restrict__ xin, const float* __restrict__ w0,
    const float* __restrict__ w1, float* __restrict__ xout) {
  __shared__ __bf16 lds_w[2][EMB * EMB];
  __shared__ float lds_t[WPB][16 * TSTRIDE];
  stage_weight(w0, lds_w[0]);
  stage_weight(w1, lds_w[1]);
  __syncthreads();

  const int wave = threadIdx.x >> 5;
  const int lane = threadIdx.x & 31;
  const int N = lane & 15;
  const int moff = (lane >> 4) << 3;
  const int stride = gridDim.x * WPB;

  for (int s = blockIdx.x * WPB + wave; s < ASTRIPS; s += stride) {
    const int row0 = s * 16;
    v16bf a[4];
    v8f cur[8];
    const float* rowp = xin + (size_t)(row0 + (lane & 15)) * EMB;
#pragma unroll
    for (int kk = 0; kk < 4; ++kk) a[kk] = lda_rowptr(rowp, kk * 32, lane, 1.0f);
#pragma unroll
    for (int n = 0; n < 8; ++n) cur[n] = zero8();
    gemm_acc(a, lds_w[0], lane, cur);
#pragma unroll
    for (int n = 0; n < 8; ++n) cur[n] = scaled_silu8(cur[n]);

    cstore_strip(cur, lds_t[wave], lane);
    wave_lds_fence();
    lda_strip(lds_t[wave], lane, a);
#pragma unroll
    for (int n = 0; n < 8; ++n) cur[n] = zero8();
    gemm_acc(a, lds_w[1], lane, cur);

#pragma unroll
    for (int n = 0; n < 8; ++n) {
      v8f y = scaled_silu8(cur[n]);
#pragma unroll
      for (int r = 0; r < 8; ++r) {
        size_t off = (size_t)(row0 + moff + r) * EMB + n * 16 + N;
        xout[off] = (xin[off] + y[r]) * INV_SQRT2;
      }
    }
  }
}

// ---------------------------------------------------------------------------
// Dense + ScaledSiLU: xout = ssilu((xin*scale) @ w)
__global__ void __launch_bounds__(BLOCK) k_dense_silu(
    const float* __restrict__ xin, const float* __restrict__ w,
    const float* __restrict__ scale_sum, float* __restrict__ xout) {
  __shared__ __bf16 lds_w[EMB * EMB];
  stage_weight(w, lds_w);
  __syncthreads();

  const int wave = threadIdx.x >> 5;
  const int lane = threadIdx.x & 31;
  const int N = lane & 15;
  const int moff = (lane >> 4) << 3;
  const int stride = gridDim.x * WPB;
  const float scl = scale_sum[0];

  for (int s = blockIdx.x * WPB + wave; s < ASTRIPS; s += stride) {
    const int row0 = s * 16;
    v16bf a[4];
    v8f acc[8];
    const float* rowp = xin + (size_t)(row0 + (lane & 15)) * EMB;
#pragma unroll
    for (int kk = 0; kk < 4; ++kk) a[kk] = lda_rowptr(rowp, kk * 32, lane, scl);
#pragma unroll
    for (int n = 0; n < 8; ++n) acc[n] = zero8();
    gemm_acc(a, lds_w, lane, acc);
#pragma unroll
    for (int n = 0; n < 8; ++n) {
      v8f y = scaled_silu8(acc[n]);
#pragma unroll
      for (int r = 0; r < 8; ++r) {
        xout[(size_t)(row0 + moff + r) * EMB + n * 16 + N] = y[r];
      }
    }
  }
}

// ---------------------------------------------------------------------------
// Edges: mlp_bf = bf@w_bf; x = hres[idx_s]*mlp_bf; atomic scatter-sum by idx_t.
__global__ void __launch_bounds__(BLOCK) k_edges(
    const float* __restrict__ bf, const int* __restrict__ idx_s,
    const int* __restrict__ idx_t, const float* __restrict__ w_bf,
    const float* __restrict__ hres, float* __restrict__ x2) {
  __shared__ __bf16 lds_w[EMB * EMB];
  stage_weight(w_bf, lds_w);
  __syncthreads();

  const int wave = threadIdx.x >> 5;
  const int lane = threadIdx.x & 31;
  const int N = lane & 15;
  const int moff = (lane >> 4) << 3;
  const int stride = gridDim.x * WPB;

  for (int s = blockIdx.x * WPB + wave; s < ESTRIPS; s += stride) {
    const int e0 = s * 16;
    // prefetch next strip of the HBM-resident bf stream (global_prefetch_b8)
    int snext = s + stride;
    if (snext < ESTRIPS) {
      __builtin_prefetch(bf + (size_t)(snext * 16 + (lane & 15)) * EMB, 0, 1);
    }

    v16bf a[4];
    const float* rowp = bf + (size_t)(e0 + (lane & 15)) * EMB;
#pragma unroll
    for (int kk = 0; kk < 4; ++kk) a[kk] = lda_rowptr(rowp, kk * 32, lane, 1.0f);
    v8f acc[8];
#pragma unroll
    for (int n = 0; n < 8; ++n) acc[n] = zero8();
    gemm_acc(a, lds_w, lane, acc);

    int si[8], ti[8];
#pragma unroll
    for (int r = 0; r < 8; ++r) {
      int e = e0 + moff + r;
      si[r] = idx_s[e];
      ti[r] = idx_t[e];
    }
#pragma unroll
    for (int n = 0; n < 8; ++n) {
#pragma unroll
      for (int r = 0; r < 8; ++r) {
        int col = n * 16 + N;
        float v = hres[(size_t)si[r] * EMB + col] * acc[n][r];
        // native non-returning global_atomic_add_f32 into L2-resident accumulator
        unsafeAtomicAdd(&x2[(size_t)ti[r] * EMB + col], v);
      }
    }
  }
}

// ---------------------------------------------------------------------------
extern "C" void kernel_launch(void* const* d_in, const int* in_sizes, int n_in,
                              void* d_out, int out_size, void* d_ws, size_t ws_size,
                              hipStream_t stream) {
  (void)in_sizes;
  (void)n_in;
  (void)out_size;
  (void)ws_size;
  const float* h = (const float*)d_in[0];
  const float* bf = (const float*)d_in[1];
  const int* idx_s = (const int*)d_in[2];
  const int* idx_t = (const int*)d_in[3];
  const float* w_bf = (const float*)d_in[4];
  const float* w_pre = (const float*)d_in[5];
  const float* w_mlp1 = (const float*)d_in[6];
  const float* w_res = (const float*)d_in[7];
  const float* scale_sum = (const float*)d_in[8];
  float* outp = (float*)d_out;

  const size_t NF = (size_t)NATOMS * EMB;
  float* hres = (float*)d_ws;   // also reused as MLP ping buffer t0
  float* x2 = hres + NF;        // scatter accumulator
  float* t1 = x2 + NF;          // MLP pong buffer

  hipMemsetAsync(x2, 0, NF * sizeof(float), stream);

  dim3 blk(BLOCK);
  const size_t WSZ = (size_t)EMB * EMB;
  // pre_residual == residual block applied to h
  k_resblock<<<AGRID, blk, 0, stream>>>(h, w_pre, w_pre + WSZ, hres);
  k_edges<<<EGRID, blk, 0, stream>>>(bf, idx_s, idx_t, w_bf, hres, x2);
  // hres consumed by k_edges; reuse it as t0
  k_dense_silu<<<AGRID, blk, 0, stream>>>(x2, w_mlp1, scale_sum, hres);
  k_resblock<<<AGRID, blk, 0, stream>>>(hres, w_res + 0 * WSZ, w_res + 1 * WSZ, t1);
  k_resblock<<<AGRID, blk, 0, stream>>>(t1, w_res + 2 * WSZ, w_res + 3 * WSZ, hres);
  k_resblock<<<AGRID, blk, 0, stream>>>(hres, w_res + 4 * WSZ, w_res + 5 * WSZ, outp);
}
